// LlamaAttention_32985348833970
// MI455X (gfx1250) — matrix operand order repaired and checked by reading, not compiled, not run on hardware
//
#include <hip/hip_runtime.h>
#include <hip/hip_bf16.h>
#include <stdint.h>

#define S_LEN   2048
#define D_MODEL 4096
#define D_HEAD  128
#define N_HEADS 32
#define N_KV    8

typedef __bf16 v16bf __attribute__((ext_vector_type(16)));
typedef __bf16 v8bf  __attribute__((ext_vector_type(8)));
typedef float  v8f   __attribute__((ext_vector_type(8)));
typedef unsigned int u32x4 __attribute__((ext_vector_type(4)));
typedef int i32x8 __attribute__((ext_vector_type(8)));
typedef int i32x4 __attribute__((ext_vector_type(4)));

// ---- CDNA5 data-mover feature detection (device pass only) -----------------
#if defined(__gfx1250__)
#  if __has_builtin(__builtin_amdgcn_tensor_load_to_lds) && __has_builtin(__builtin_amdgcn_s_wait_tensorcnt)
#    define USE_TDM 1
#  endif
#  if __has_builtin(__builtin_amdgcn_global_load_async_to_lds_b128) && __has_builtin(__builtin_amdgcn_s_wait_asynccnt)
#    define USE_ASYNC 1
#  endif
#endif
#ifndef USE_TDM
#define USE_TDM 0
#endif
#ifndef USE_ASYNC
#define USE_ASYNC 0
#endif

#if USE_ASYNC
// GLOBAL_LOAD_ASYNC_TO_LDS_B128: 16B global -> LDS, tracked by ASYNCcnt.
__device__ __forceinline__ void async_b128(const void* g, void* l) {
    __builtin_amdgcn_global_load_async_to_lds_b128((i32x4*)g, (i32x4*)l, 0, 0);
}
#endif

#if USE_TDM
// TENSOR_LOAD_TO_LDS with a 2D D# (ISA 8.3/8.4): bf16 elements (data_size=1 -> 2B),
// tile (tile0 x tile1) of a row-major [dim1 x dim0] tensor, packed row-major into LDS.
__device__ __forceinline__ void tdm_load_2d(unsigned int lds_off, const void* gaddr,
                                            unsigned int dim0, unsigned int dim1,
                                            unsigned int stride0,
                                            unsigned int tile0, unsigned int tile1) {
    unsigned long long ga = (unsigned long long)(uintptr_t)gaddr;
    u32x4 g0;
    g0[0] = 1u;                                            // count=1, user mode
    g0[1] = lds_off;                                       // lds_addr
    g0[2] = (unsigned int)ga;                              // global_addr[31:0]
    g0[3] = (unsigned int)((ga >> 32) & 0x1FFFFFFull) | (2u << 30);  // addr[56:32] | type=2
    i32x8 g1;
    g1[0] = (int)(1u << 16);                               // data_size=1 (2 bytes)
    g1[1] = (int)((dim0 & 0xFFFFu) << 16);                 // tensor_dim0[15:0]
    g1[2] = (int)((dim0 >> 16) | ((dim1 & 0xFFFFu) << 16));// dim0[31:16] | dim1[15:0]
    g1[3] = (int)(((dim1 >> 16) & 0xFFFFu) | (tile0 << 16));// dim1[31:16] | tile_dim0
    g1[4] = (int)(tile1 & 0xFFFFu);                        // tile_dim1 (tile_dim2=0)
    g1[5] = (int)stride0;                                  // tensor_dim0_stride[31:0]
    g1[6] = 0;                                             // stride[47:32], dim1_stride lo
    g1[7] = 0;
    i32x4 z4 = {0, 0, 0, 0};
    i32x8 z8 = {0, 0, 0, 0, 0, 0, 0, 0};
    __builtin_amdgcn_tensor_load_to_lds(g0, g1, z4, z4, z8, 0);
}
#endif

__device__ __forceinline__ v16bf combine16(v8bf lo, v8bf hi) {
    return __builtin_shufflevector(lo, hi, 0,1,2,3,4,5,6,7,8,9,10,11,12,13,14,15);
}

__device__ __forceinline__ float hmax16(float v) {
    v = fmaxf(v, __shfl_xor(v, 1, 32));
    v = fmaxf(v, __shfl_xor(v, 2, 32));
    v = fmaxf(v, __shfl_xor(v, 4, 32));
    v = fmaxf(v, __shfl_xor(v, 8, 32));
    return v;
}
__device__ __forceinline__ float hsum16(float v) {
    v += __shfl_xor(v, 1, 32);
    v += __shfl_xor(v, 2, 32);
    v += __shfl_xor(v, 4, 32);
    v += __shfl_xor(v, 8, 32);
    return v;
}

// ---------------------------------------------------------------- fp32->bf16
__global__ __launch_bounds__(256) void cvt_bf16_kernel(const float* __restrict__ in,
                                                       unsigned short* __restrict__ out_, int n) {
    __bf16* out = (__bf16*)out_;
    int i = blockIdx.x * blockDim.x + threadIdx.x;
    if (i < n) out[i] = (__bf16)in[i];
}

// ---------------------------------------------------------------- RMSNorm
__global__ __launch_bounds__(256) void rmsnorm_kernel(const float* __restrict__ x,
                                                      const float* __restrict__ w,
                                                      unsigned short* __restrict__ h_) {
    __bf16* h = (__bf16*)h_;
    int row = blockIdx.x, t = threadIdx.x;
    const float* xr = x + (size_t)row * D_MODEL;
    float acc = 0.f;
    for (int i = t; i < D_MODEL; i += 256) { float v = xr[i]; acc += v * v; }
    acc += __shfl_xor(acc, 1, 32); acc += __shfl_xor(acc, 2, 32);
    acc += __shfl_xor(acc, 4, 32); acc += __shfl_xor(acc, 8, 32);
    acc += __shfl_xor(acc, 16, 32);
    __shared__ float red[8];
    if ((t & 31) == 0) red[t >> 5] = acc;
    __syncthreads();
    float tot = 0.f;
#pragma unroll
    for (int i = 0; i < 8; i++) tot += red[i];
    float scale = rsqrtf(tot * (1.0f / D_MODEL) + 1e-5f);
    for (int i = t; i < D_MODEL; i += 256)
        h[(size_t)row * D_MODEL + i] = (__bf16)(xr[i] * scale * w[i]);
}

// ---------------------------------------------------------------- NT GEMM
// C[M,N] = A[M,K] * B[N,K]^T (+ residual), A/B bf16 row-major, C fp32.
// 256 thr (8 waves, 4x2 wave grid), block tile 128x128, per-wave 32x64,
// K-step 64, double-buffered LDS, TDM descriptor tile loads
// (fallback: async-to-LDS, then sync staging).
__global__ __launch_bounds__(256) void gemm_nt_kernel(const unsigned short* __restrict__ A_,
                                                      const unsigned short* __restrict__ B_,
                                                      const float* __restrict__ res,
                                                      float* __restrict__ C,
                                                      int M, int N, int K) {
    const __bf16* A = (const __bf16*)A_;
    const __bf16* B = (const __bf16*)B_;
    __shared__ alignas(128) __bf16 Als[2][128][64];
    __shared__ alignas(128) __bf16 Bls[2][128][64];

    const int tid = threadIdx.x, lane = tid & 31, w = tid >> 5;
    const int l16 = lane & 15, hi16 = lane >> 4;
    const int mbase = blockIdx.y * 128, nbase = blockIdx.x * 128;
    const int rbase = (w >> 1) * 32;       // wave row tile (4 waves along M)
    const int cbase = (w & 1) * 64;        // wave col tile (2 waves along N)

    v8f c[2][4];
    v8f z = {};
#pragma unroll
    for (int im = 0; im < 2; im++)
#pragma unroll
        for (int nt = 0; nt < 4; nt++) c[im][nt] = z;

    auto stage = [&](int buf, int kk) {
#if USE_TDM
        if (w == 0) {
            tdm_load_2d((unsigned int)(uintptr_t)&Als[buf][0][0],
                        A + (size_t)mbase * K + kk,
                        (unsigned)K, (unsigned)M, (unsigned)K, 64u, 128u);
            tdm_load_2d((unsigned int)(uintptr_t)&Bls[buf][0][0],
                        B + (size_t)nbase * K + kk,
                        (unsigned)K, (unsigned)N, (unsigned)K, 64u, 128u);
        }
#elif USE_ASYNC
        const __bf16* ga = A + (size_t)(mbase + (tid >> 1)) * K + kk + (tid & 1) * 32;
        __bf16* la = &Als[buf][tid >> 1][(tid & 1) * 32];
        async_b128(ga, la);           async_b128(ga + 8, la + 8);
        async_b128(ga + 16, la + 16); async_b128(ga + 24, la + 24);
        const __bf16* gb = B + (size_t)(nbase + (tid >> 1)) * K + kk + (tid & 1) * 32;
        __bf16* lb = &Bls[buf][tid >> 1][(tid & 1) * 32];
        async_b128(gb, lb);           async_b128(gb + 8, lb + 8);
        async_b128(gb + 16, lb + 16); async_b128(gb + 24, lb + 24);
#else
        const __bf16* ga = A + (size_t)(mbase + (tid >> 1)) * K + kk + (tid & 1) * 32;
        __bf16* la = &Als[buf][tid >> 1][(tid & 1) * 32];
        *(v16bf*)la        = *(const v16bf*)ga;
        *(v16bf*)(la + 16) = *(const v16bf*)(ga + 16);
        const __bf16* gb = B + (size_t)(nbase + (tid >> 1)) * K + kk + (tid & 1) * 32;
        __bf16* lb = &Bls[buf][tid >> 1][(tid & 1) * 32];
        *(v16bf*)lb        = *(const v16bf*)gb;
        *(v16bf*)(lb + 16) = *(const v16bf*)(gb + 16);
        if (kk + 64 < K) {
            __builtin_prefetch(ga + 64, 0, 1);
            __builtin_prefetch(gb + 64, 0, 1);
        }
#endif
    };

    stage(0, 0);
    int buf = 0;
    for (int kk = 0; kk < K; kk += 64) {
        const bool more = (kk + 64) < K;
        if (more) stage(buf ^ 1, kk + 64);
#if USE_TDM
        if (w == 0) {
            if (more) __builtin_amdgcn_s_wait_tensorcnt(2);
            else      __builtin_amdgcn_s_wait_tensorcnt(0);
        }
#elif USE_ASYNC
        if (more) __builtin_amdgcn_s_wait_asynccnt(8);
        else      __builtin_amdgcn_s_wait_asynccnt(0);
#endif
        __syncthreads();
#pragma unroll
        for (int ks = 0; ks < 2; ks++) {
            v16bf af[2];
#pragma unroll
            for (int im = 0; im < 2; im++) {
                v8bf alo = *(const v8bf*)&Als[buf][rbase + im * 16 + l16][ks * 32 + hi16 * 16];
                v8bf ahi = *(const v8bf*)&Als[buf][rbase + im * 16 + l16][ks * 32 + hi16 * 16 + 8];
                af[im] = combine16(alo, ahi);
            }
#pragma unroll
            for (int nt = 0; nt < 4; nt++) {
                v16bf bfr = *(const v16bf*)&Bls[buf][cbase + nt * 16 + l16][ks * 32 + hi16 * 16];
#pragma unroll
                for (int im = 0; im < 2; im++)
                    c[im][nt] = __builtin_amdgcn_wmma_f32_16x16x32_bf16(false, af[im], false, bfr,
                                                                        (short)0, c[im][nt], false, false);
            }
        }
        __syncthreads();
        buf ^= 1;
    }
#pragma unroll
    for (int im = 0; im < 2; im++)
#pragma unroll
        for (int nt = 0; nt < 4; nt++) {
#pragma unroll
            for (int j = 0; j < 8; j++) {
                int r = mbase + rbase + im * 16 + hi16 * 8 + j;
                int col = nbase + cbase + nt * 16 + l16;
                float v = c[im][nt][j];
                if (res) v += res[(size_t)r * N + col];
                C[(size_t)r * N + col] = v;
            }
        }
}

// ---------------------------------------------------------------- RoPE + pack
__global__ __launch_bounds__(256) void rope_q_kernel(const float* __restrict__ qf,
                                                     const float* __restrict__ cs,
                                                     const float* __restrict__ sn,
                                                     unsigned short* __restrict__ Qb_) {
    __bf16* Qb = (__bf16*)Qb_;
    int idx = blockIdx.x * blockDim.x + threadIdx.x;      // pair index
    if (idx >= S_LEN * (N_HEADS * D_HEAD / 2)) return;
    int s = idx >> 11, j = idx & 2047;
    int col = j * 2, h = col >> 7, d = col & 127;
    float c = cs[s * D_HEAD + d], si = sn[s * D_HEAD + d];
    float x0 = qf[(size_t)s * (N_HEADS * D_HEAD) + col];
    float x1 = qf[(size_t)s * (N_HEADS * D_HEAD) + col + 1];
    const float sc = 0.08838834764831845f;                // 1/sqrt(128) folded into Q
    __bf16* dst = Qb + ((size_t)(h * S_LEN + s)) * D_HEAD + d;
    dst[0] = (__bf16)((x0 * c - x1 * si) * sc);
    dst[1] = (__bf16)((x1 * c + x0 * si) * sc);
}

__global__ __launch_bounds__(256) void rope_k_kernel(const float* __restrict__ kf,
                                                     const float* __restrict__ cs,
                                                     const float* __restrict__ sn,
                                                     unsigned short* __restrict__ Kb_) {
    __bf16* Kb = (__bf16*)Kb_;
    int idx = blockIdx.x * blockDim.x + threadIdx.x;
    if (idx >= S_LEN * (N_KV * D_HEAD / 2)) return;
    int s = idx >> 9, j = idx & 511;
    int col = j * 2, g = col >> 7, d = col & 127;
    float c = cs[s * D_HEAD + d], si = sn[s * D_HEAD + d];
    float x0 = kf[(size_t)s * (N_KV * D_HEAD) + col];
    float x1 = kf[(size_t)s * (N_KV * D_HEAD) + col + 1];
    __bf16* dst = Kb + ((size_t)(g * S_LEN + s)) * D_HEAD + d;
    dst[0] = (__bf16)(x0 * c - x1 * si);
    dst[1] = (__bf16)(x1 * c + x0 * si);
}

// V: [S, 1024] f32 -> Vt[kv*128 + d][S] bf16, LDS-tiled transpose.
__global__ __launch_bounds__(256) void pack_v_kernel(const float* __restrict__ vf,
                                                     unsigned short* __restrict__ Vt_) {
    __bf16* Vt = (__bf16*)Vt_;
    __shared__ __bf16 tile[32][33];
    int tx = threadIdx.x & 31, ty = threadIdx.x >> 5;
    int s0 = blockIdx.x * 32, c0 = blockIdx.y * 32;
#pragma unroll
    for (int i = 0; i < 32; i += 8)
        tile[ty + i][tx] = (__bf16)vf[(size_t)(s0 + ty + i) * (N_KV * D_HEAD) + c0 + tx];
    __syncthreads();
#pragma unroll
    for (int i = 0; i < 32; i += 8)
        Vt[(size_t)(c0 + ty + i) * S_LEN + s0 + tx] = tile[tx][ty + i];
}

// ---------------------------------------------------------------- Flash attention
// grid (S/128, N_HEADS); 256 thr = 8 waves; wave = 16 q-rows x d=128.
// 32-wide causal K-blocks, double-buffered LDS, async-to-LDS staging,
// online softmax, WMMA for Q*K^T and P*V.
__global__ __launch_bounds__(256) void attn_kernel(const unsigned short* __restrict__ Qb_,
                                                   const unsigned short* __restrict__ Kb_,
                                                   const unsigned short* __restrict__ Vt_,
                                                   unsigned short* __restrict__ Ao_) {
    const __bf16* Qb = (const __bf16*)Qb_;
    const __bf16* Kb = (const __bf16*)Kb_;
    const __bf16* Vt = (const __bf16*)Vt_;
    __bf16* Ao = (__bf16*)Ao_;

    __shared__ alignas(128) __bf16 Kls[2][32][128];   // [t][d]
    __shared__ alignas(128) __bf16 Vls[2][128][32];   // [d][t]
    __shared__ alignas(128) __bf16 Pls[8][16][32];    // per-wave P staging

    const int tid = threadIdx.x, lane = tid & 31, w = tid >> 5;
    const int l16 = lane & 15, hi16 = lane >> 4;
    const int h = blockIdx.y, gh = h >> 2;
    const int qbase = blockIdx.x * 128;
    const int qb = qbase + w * 16;

    // Q fragments (A-layout), 16 rows x 128 d
    v16bf qfr[4];
    {
        const __bf16* base = Qb + ((size_t)(h * S_LEN + qb + l16)) * D_HEAD + hi16 * 16;
#pragma unroll
        for (int d0 = 0; d0 < 4; d0++) {
            v8bf lo = *(const v8bf*)(base + d0 * 32);
            v8bf hi = *(const v8bf*)(base + d0 * 32 + 8);
            qfr[d0] = combine16(lo, hi);
        }
    }

    v8f o[8];
    float mi[8], li[8];
    v8f z = {};
#pragma unroll
    for (int j = 0; j < 8; j++) { o[j] = z; mi[j] = -1e30f; li[j] = 0.f; }

    auto stageKV = [&](int buf, int tb) {
        const __bf16* gk = Kb + ((size_t)(gh * S_LEN + tb + (tid >> 3))) * D_HEAD + (tid & 7) * 16;
        __bf16* lk = &Kls[buf][tid >> 3][(tid & 7) * 16];
        const __bf16* gv = Vt + ((size_t)(gh * D_HEAD + (tid >> 1))) * S_LEN + tb + (tid & 1) * 16;
        __bf16* lv = &Vls[buf][tid >> 1][(tid & 1) * 16];
#if USE_ASYNC
        async_b128(gk, lk); async_b128(gk + 8, lk + 8);
        async_b128(gv, lv); async_b128(gv + 8, lv + 8);
#else
        *(v16bf*)lk = *(const v16bf*)gk;
        *(v16bf*)lv = *(const v16bf*)gv;
#endif
    };

    const int nb = (qbase + 128) >> 5;
    stageKV(0, 0);
    int buf = 0;
    for (int b = 0; b < nb; b++) {
        const int tb = b * 32;
        const bool more = (b + 1) < nb;
        if (more) stageKV(buf ^ 1, tb + 32);
#if USE_ASYNC
        if (more) __builtin_amdgcn_s_wait_asynccnt(4);
        else      __builtin_amdgcn_s_wait_asynccnt(0);
#endif
        __syncthreads();

        // scores: two 16x16 tiles over the 32-wide block
        v8f s0 = z, s1 = z;
#pragma unroll
        for (int d0 = 0; d0 < 4; d0++) {
            v16bf b0 = *(const v16bf*)&Kls[buf][l16][d0 * 32 + hi16 * 16];
            v16bf b1 = *(const v16bf*)&Kls[buf][16 + l16][d0 * 32 + hi16 * 16];
            s0 = __builtin_amdgcn_wmma_f32_16x16x32_bf16(false, qfr[d0], false, b0, (short)0, s0, false, false);
            s1 = __builtin_amdgcn_wmma_f32_16x16x32_bf16(false, qfr[d0], false, b1, (short)0, s1, false, false);
        }

        // online softmax per row (each row lives within one 16-lane half)
        float alpha[8];
#pragma unroll
        for (int j = 0; j < 8; j++) {
            int rg = qb + hi16 * 8 + j;
            float a0 = (tb + l16      <= rg) ? s0[j] : -1e30f;
            float a1 = (tb + 16 + l16 <= rg) ? s1[j] : -1e30f;
            float mt = hmax16(fmaxf(a0, a1));
            float mn = fmaxf(mi[j], mt);
            float p0 = __expf(a0 - mn), p1 = __expf(a1 - mn);
            float al = __expf(mi[j] - mn);
            li[j] = li[j] * al + hsum16(p0 + p1);
            mi[j] = mn;
            alpha[j] = al;
            int r = hi16 * 8 + j;
            Pls[w][r][l16]      = (__bf16)p0;
            Pls[w][r][16 + l16] = (__bf16)p1;
        }
#pragma unroll
        for (int dt = 0; dt < 8; dt++) {
            v8f t = o[dt];
#pragma unroll
            for (int j = 0; j < 8; j++) t[j] *= alpha[j];
            o[dt] = t;
        }

        // re-fragment P (16x32) as A-matrix and accumulate P*V
        v8bf plo = *(const v8bf*)&Pls[w][l16][hi16 * 16];
        v8bf phi = *(const v8bf*)&Pls[w][l16][hi16 * 16 + 8];
        v16bf pf = combine16(plo, phi);
#pragma unroll
        for (int dt = 0; dt < 8; dt++) {
            v16bf bv = *(const v16bf*)&Vls[buf][dt * 16 + l16][hi16 * 16];
            o[dt] = __builtin_amdgcn_wmma_f32_16x16x32_bf16(false, pf, false, bv, (short)0, o[dt], false, false);
        }
        __syncthreads();
        buf ^= 1;
    }

    float linv[8];
#pragma unroll
    for (int j = 0; j < 8; j++) linv[j] = 1.0f / li[j];
#pragma unroll
    for (int dt = 0; dt < 8; dt++) {
#pragma unroll
        for (int j = 0; j < 8; j++) {
            int r = qb + hi16 * 8 + j;
            int c = h * D_HEAD + dt * 16 + l16;
            Ao[(size_t)r * (N_HEADS * D_HEAD) + c] = (__bf16)(o[dt][j] * linv[j]);
        }
    }
}

// ---------------------------------------------------------------- host
extern "C" void kernel_launch(void* const* d_in, const int* in_sizes, int n_in,
                              void* d_out, int out_size, void* d_ws, size_t ws_size,
                              hipStream_t stream) {
    (void)in_sizes; (void)n_in; (void)out_size; (void)ws_size;
    const float* x     = (const float*)d_in[0];
    const float* r_cos = (const float*)d_in[1];
    const float* r_sin = (const float*)d_in[2];
    // d_in[3] = mask: causality computed analytically in attn_kernel
    const float* nw    = (const float*)d_in[4];
    const float* wq    = (const float*)d_in[5];
    const float* wk    = (const float*)d_in[6];
    const float* wv    = (const float*)d_in[7];
    const float* wo    = (const float*)d_in[8];
    float* out = (float*)d_out;

    char* ws = (char*)d_ws;
    size_t off = 0;
    auto alloc = [&](size_t bytes) -> void* {
        void* p = ws + off;
        off += (bytes + 255) & ~(size_t)255;
        return p;
    };
    const size_t NQ = (size_t)D_MODEL * D_MODEL;
    const size_t NK = (size_t)(N_KV * D_HEAD) * D_MODEL;
    unsigned short* wq_b  = (unsigned short*)alloc(NQ * 2);
    unsigned short* wk_b  = (unsigned short*)alloc(NK * 2);
    unsigned short* wv_b  = (unsigned short*)alloc(NK * 2);
    unsigned short* wo_b  = (unsigned short*)alloc(NQ * 2);
    unsigned short* h_b   = (unsigned short*)alloc((size_t)S_LEN * D_MODEL * 2);
    float*          qf    = (float*)alloc((size_t)S_LEN * D_MODEL * 4);
    float*          kf    = (float*)alloc((size_t)S_LEN * N_KV * D_HEAD * 4);
    float*          vf    = (float*)alloc((size_t)S_LEN * N_KV * D_HEAD * 4);
    unsigned short* Qb    = (unsigned short*)alloc((size_t)S_LEN * D_MODEL * 2);
    unsigned short* Kb    = (unsigned short*)alloc((size_t)S_LEN * N_KV * D_HEAD * 2);
    unsigned short* Vt    = (unsigned short*)alloc((size_t)S_LEN * N_KV * D_HEAD * 2);
    unsigned short* att_b = (unsigned short*)alloc((size_t)S_LEN * D_MODEL * 2);

    // 1) weights -> bf16
    cvt_bf16_kernel<<<(int)((NQ + 255) / 256), 256, 0, stream>>>(wq, wq_b, (int)NQ);
    cvt_bf16_kernel<<<(int)((NK + 255) / 256), 256, 0, stream>>>(wk, wk_b, (int)NK);
    cvt_bf16_kernel<<<(int)((NK + 255) / 256), 256, 0, stream>>>(wv, wv_b, (int)NK);
    cvt_bf16_kernel<<<(int)((NQ + 255) / 256), 256, 0, stream>>>(wo, wo_b, (int)NQ);

    // 2) RMSNorm
    rmsnorm_kernel<<<S_LEN, 256, 0, stream>>>(x, nw, h_b);

    // 3) Q/K/V projections (NT GEMM, bf16 WMMA, fp32 out)
    dim3 gq(D_MODEL / 128, S_LEN / 128);
    dim3 gk((N_KV * D_HEAD) / 128, S_LEN / 128);
    gemm_nt_kernel<<<gq, 256, 0, stream>>>(h_b, wq_b, nullptr, qf, S_LEN, D_MODEL, D_MODEL);
    gemm_nt_kernel<<<gk, 256, 0, stream>>>(h_b, wk_b, nullptr, kf, S_LEN, N_KV * D_HEAD, D_MODEL);
    gemm_nt_kernel<<<gk, 256, 0, stream>>>(h_b, wv_b, nullptr, vf, S_LEN, N_KV * D_HEAD, D_MODEL);

    // 4) RoPE + layout packing
    rope_q_kernel<<<(S_LEN * N_HEADS * D_HEAD / 2) / 256, 256, 0, stream>>>(qf, r_cos, r_sin, Qb);
    rope_k_kernel<<<(S_LEN * N_KV * D_HEAD / 2) / 256, 256, 0, stream>>>(kf, r_cos, r_sin, Kb);
    pack_v_kernel<<<dim3(S_LEN / 32, (N_KV * D_HEAD) / 32), 256, 0, stream>>>(vf, Vt);

    // 5) flash attention
    attn_kernel<<<dim3(S_LEN / 128, N_HEADS), 256, 0, stream>>>(Qb, Kb, Vt, att_b);

    // 6) output projection + residual (fp32 out)
    gemm_nt_kernel<<<gq, 256, 0, stream>>>(att_b, wo_b, x, out, S_LEN, D_MODEL, D_MODEL);
}
